// BB_ShortestPathNet_1829656068357
// MI455X (gfx1250) — matrix operand order repaired and checked by reading, not compiled, not run on hardware
//
#include <hip/hip_runtime.h>

typedef __attribute__((ext_vector_type(16))) _Float16 v16h;
typedef __attribute__((ext_vector_type(8)))  float    v8f;
typedef unsigned int v4ui __attribute__((ext_vector_type(4)));
typedef int          v8i  __attribute__((ext_vector_type(8)));
typedef int          v4i  __attribute__((ext_vector_type(4)));

#define BB_INF 1.0e10f

#if __has_builtin(__builtin_amdgcn_tensor_load_to_lds) && \
    __has_builtin(__builtin_amdgcn_s_wait_tensorcnt)
#define USE_TDM 1
#else
#define USE_TDM 0
#endif

// ---------------------------------------------------------------------------
// f32 -> f16 conversion (grid-stride)
// ---------------------------------------------------------------------------
__global__ void f32_to_f16_kernel(const float* __restrict__ src,
                                  _Float16* __restrict__ dst, long n) {
  long i = (long)blockIdx.x * blockDim.x + threadIdx.x;
  long stride = (long)gridDim.x * blockDim.x;
  for (; i < n; i += stride) dst[i] = (_Float16)src[i];
}

struct alignas(16) H8 { _Float16 v[8]; };

#if USE_TDM
// ---------------------------------------------------------------------------
// TDM: DMA a 2D tile (tileRows x tileK f16 elements, row stride = strideElems)
// from global to LDS at ldsByte, padding +4 DWORDs after every 16 DWORDs so
// the LDS row stride becomes 40 halves (80B). OOB rows/cols read as zero.
// D# packing per CDNA5 ISA ch.8 (group0 128b, group1 256b; groups 2/3 unused).
// Toolchain uses the 6-arg builtin form:
//   (uint32x4 g0, int32x8 g1, int32x4, int32x4, int32x8, i32 cpol)
// ---------------------------------------------------------------------------
__device__ __forceinline__ void tdm_load_2d(const void* gtile, unsigned ldsByte,
                                            unsigned long long remRows,
                                            unsigned long long remK,
                                            unsigned long long strideElems,
                                            unsigned tileRows, unsigned tileK) {
  const unsigned long long ga = (unsigned long long)(size_t)gtile;
  v4ui g0;
  g0.x = 1u;                                           // count=1, user D#
  g0.y = ldsByte;                                      // lds_addr
  g0.z = (unsigned)ga;                                 // global_addr[31:0]
  g0.w = (unsigned)((ga >> 32) & 0x01FFFFFFull) | (2u << 30);  // [56:32] | type=2

  union { unsigned long long q[4]; v8i v; } g1;
  // mask=0 | data_size=1(2B) | pad_enable | pad_interval=3(16DW) | pad_amount=3(4DW)
  const unsigned long long flags =
      (1ull << 16) | (1ull << 20) | (3ull << 22) | (3ull << 25);
  g1.q[0] = flags | ((remK & 0xFFFFull) << 48);                         // dim0[15:0]
  g1.q[1] = ((remK >> 16) & 0xFFFFull)                                  // dim0[31:16]
          | ((remRows & 0xFFFFFFFFull) << 16)                           // dim1
          | ((unsigned long long)tileK << 48);                          // tile_dim0
  g1.q[2] = (unsigned long long)tileRows                                // tile_dim1
          | ((strideElems & 0xFFFFFFFFull) << 32);                      // stride0[31:0]
  g1.q[3] = (strideElems >> 32) & 0xFFFFull;                            // stride0[47:32]

  const v4i gz4 = {0, 0, 0, 0};
  const v8i gz8 = {0, 0, 0, 0, 0, 0, 0, 0};
  __builtin_amdgcn_tensor_load_to_lds(g0, g1.v, gz4, gz4, gz8, 0);
}
#endif

// ---------------------------------------------------------------------------
// C[M,N] = act(A[M,K] @ B[N,K]^T + bias)   (both operands row-major, NT GEMM)
// Block: 256 threads = 8 waves. Block tile 64(M) x 128(N). K-step 32.
// Wave (wm 0..3, wn 0..1) computes a 16x64 tile = 4 x v_wmma_f32_16x16x32_f16.
// TDM path: double-buffered LDS, one barrier per K-step, DMA prefetch.
// ---------------------------------------------------------------------------
template<bool RELU, bool OUT_F16>
__global__ __launch_bounds__(256) void gemm_f16_wmma(
    const _Float16* __restrict__ A,   // M x K
    const _Float16* __restrict__ B,   // N x K
    const float*    __restrict__ bias,// N
    void*           __restrict__ Cout,// M x N  (f16 or f32)
    int Mdim, int Ndim, int Kdim)
{
  constexpr int LDT = 40;               // LDS row stride in halves (80B, 16B-aligned)
  __shared__ _Float16 As[2][64 * LDT];
  __shared__ _Float16 Bs[2][128 * LDT];

  const int tid  = threadIdx.x;
  const int lane = tid & 31;
  const int wid  = tid >> 5;
  const int wm   = wid & 3;             // M sub-tile 0..3
  const int wn   = wid >> 2;            // N half 0..1
  const int mBase = blockIdx.y * 64;
  const int nBase = blockIdx.x * 128;

  v8f acc[4];
#pragma unroll
  for (int s = 0; s < 4; ++s)
#pragma unroll
    for (int r = 0; r < 8; ++r) acc[s][r] = 0.0f;

  const int nSteps = Kdim / 32;

#if USE_TDM
  const unsigned aLds[2] = {(unsigned)(size_t)&As[0][0], (unsigned)(size_t)&As[1][0]};
  const unsigned bLds[2] = {(unsigned)(size_t)&Bs[0][0], (unsigned)(size_t)&Bs[1][0]};
  if (wid == 0) {
    tdm_load_2d(A + (long)mBase * Kdim, aLds[0],
                (unsigned long long)(Mdim - mBase), (unsigned long long)Kdim,
                (unsigned long long)Kdim, 64u, 32u);
    tdm_load_2d(B + (long)nBase * Kdim, bLds[0],
                (unsigned long long)(Ndim - nBase), (unsigned long long)Kdim,
                (unsigned long long)Kdim, 128u, 32u);
  }
#else
  // manual staging indices: one 16B chunk of A, two of B, per thread per step
  const int sRow = tid >> 2;            // 0..63
  const int sK   = (tid & 3) * 8;       // 0,8,16,24
#endif

  for (int i = 0; i < nSteps; ++i) {
    const int buf = i & 1;

#if USE_TDM
    if (wid == 0) __builtin_amdgcn_s_wait_tensorcnt(0);
    __syncthreads();                    // tile[buf] ready; tile[buf^1] fully read
    if (i + 1 < nSteps && wid == 0) {
      const int k1 = (i + 1) * 32;
      tdm_load_2d(A + (long)mBase * Kdim + k1, aLds[buf ^ 1],
                  (unsigned long long)(Mdim - mBase),
                  (unsigned long long)(Kdim - k1),
                  (unsigned long long)Kdim, 64u, 32u);
      tdm_load_2d(B + (long)nBase * Kdim + k1, bLds[buf ^ 1],
                  (unsigned long long)(Ndim - nBase),
                  (unsigned long long)(Kdim - k1),
                  (unsigned long long)Kdim, 128u, 32u);
    }
#else
    const int k0 = i * 32;
    __syncthreads();
    { // A tile: 64 rows x 32 halves
      const long gr = mBase + sRow;
      const uint4 val = *(const uint4*)(A + gr * (long)Kdim + k0 + sK);
      *(uint4*)(&As[0][sRow * LDT + sK]) = val;
    }
#pragma unroll
    for (int p = 0; p < 2; ++p) { // B tile: 128 rows x 32 halves
      const int br = sRow + p * 64;
      int gn = nBase + br;
      if (gn > Ndim - 1) gn = Ndim - 1;  // clamp for N not multiple of 128
      const uint4 val = *(const uint4*)(B + (long)gn * Kdim + k0 + sK);
      *(uint4*)(&Bs[0][br * LDT + sK]) = val;
    }
    __syncthreads();
#endif

    const _Float16* __restrict__ Asb = &As[USE_TDM ? buf : 0][0];
    const _Float16* __restrict__ Bsb = &Bs[USE_TDM ? buf : 0][0];

    // A fragment: row = lane&15; elems 0..7 -> K kb..kb+7, 8..15 -> K kb+16..kb+23
    const int ar  = wm * 16 + (lane & 15);
    const int akb = (lane >> 4) * 8;
    const H8 alo = *(const H8*)(&Asb[ar * LDT + akb]);
    const H8 ahi = *(const H8*)(&Asb[ar * LDT + akb + 16]);
    v16h afrag;
#pragma unroll
    for (int e = 0; e < 8; ++e) { afrag[e] = alo.v[e]; afrag[e + 8] = ahi.v[e]; }

    // B fragments: col = lane&15; elems 0..7 -> K kb..kb+7, 8..15 -> K kb+16..kb+23
    const int bkb = (lane >> 4) * 8;
#pragma unroll
    for (int s = 0; s < 4; ++s) {
      const int bc = wn * 64 + s * 16 + (lane & 15);
      const H8 blo = *(const H8*)(&Bsb[bc * LDT + bkb]);
      const H8 bhi = *(const H8*)(&Bsb[bc * LDT + bkb + 16]);
      v16h bfrag;
#pragma unroll
      for (int e = 0; e < 8; ++e) { bfrag[e] = blo.v[e]; bfrag[e + 8] = bhi.v[e]; }
      acc[s] = __builtin_amdgcn_wmma_f32_16x16x32_f16(
          /*neg_a=*/false, afrag, /*neg_b=*/false, bfrag,
          /*c_mod=*/(short)0, acc[s], /*reuse_a=*/false, /*reuse_b=*/false);
    }
  }

  // epilogue: C/D layout -> col = lane&15, row = r + (lane>>4)*8
  const int rowBase = mBase + wm * 16 + ((lane >> 4) * 8);
#pragma unroll
  for (int s = 0; s < 4; ++s) {
    const int col = nBase + wn * 64 + s * 16 + (lane & 15);
    if (col < Ndim) {
      const float bv = bias[col];
#pragma unroll
      for (int r = 0; r < 8; ++r) {
        float v = acc[s][r] + bv;
        if (RELU) v = v > 0.0f ? v : 0.1f * v;   // leaky_relu(0.1)
        const long idx = (long)(rowBase + r) * Ndim + col;
        if (OUT_F16) ((_Float16*)Cout)[idx] = (_Float16)v;
        else         ((float*)Cout)[idx]    = v;
      }
    }
  }
}

// ---------------------------------------------------------------------------
// Per-batch 20x20 Bellman-Ford (exactly 400 Jacobi sweeps, negative weights OK)
// + backtrack from (19,19) following first-argmin neighbor, 0/1 path output.
// One 256-thread block per batch item.
// ---------------------------------------------------------------------------
__global__ __launch_bounds__(256) void bf_solver(const float* __restrict__ wgrid,
                                                 float* __restrict__ out) {
  constexpr int P = 24;                 // padded row stride (22 rows used)
  const int b = blockIdx.x;
  const int tid = threadIdx.x;

  __shared__ float bufA[22 * P];
  __shared__ float bufB[22 * P];
  __shared__ float wl[400];
  __shared__ unsigned char pathm[400];

  for (int i = tid; i < 22 * P; i += 256) { bufA[i] = BB_INF; bufB[i] = BB_INF; }
  for (int c = tid; c < 400; c += 256) {
    wl[c] = wgrid[(long)b * 400 + c];
    pathm[c] = 0;
  }
  __syncthreads();
  if (tid == 0) bufA[1 * P + 1] = wl[0];   // dist0[0,0] = w[0,0]
  __syncthreads();

  float* cur = bufA;
  float* nxt = bufB;
  for (int it = 0; it < 400; ++it) {       // exactly M*M sweeps like the reference
    for (int c = tid; c < 400; c += 256) {
      const int i = c / 20, j = c % 20;
      const int base = (i + 1) * P + (j + 1);
      float m;
      m =          cur[base - P - 1];
      m = fminf(m, cur[base - P    ]);
      m = fminf(m, cur[base - P + 1]);
      m = fminf(m, cur[base     - 1]);
      m = fminf(m, cur[base     + 1]);
      m = fminf(m, cur[base + P - 1]);
      m = fminf(m, cur[base + P    ]);
      m = fminf(m, cur[base + P + 1]);
      float v = fminf(cur[base], wl[c] + m);
      if (c == 0) v = wl[0];               // .at[0,0].set(w[0,0]) each sweep
      nxt[base] = v;
    }
    __syncthreads();
    float* t = cur; cur = nxt; nxt = t;
  }

  if (tid == 0) {                          // backtrack (matches jnp.argmin order)
    const int di[8] = {-1,-1,-1, 0, 0, 1, 1, 1};
    const int dj[8] = {-1, 0, 1,-1, 1,-1, 0, 1};
    int i = 19, j = 19;
    bool done = false;
    for (int step = 0; step < 400; ++step) {
      pathm[i * 20 + j] = 1;
      done = done || (i == 0 && j == 0);
      float best = cur[(i + di[0] + 1) * P + (j + dj[0] + 1)];
      int bk = 0;
      for (int k = 1; k < 8; ++k) {
        const float cand = cur[(i + di[k] + 1) * P + (j + dj[k] + 1)];
        if (cand < best) { best = cand; bk = k; }  // first-min like argmin
      }
      if (!done) { i += di[bk]; j += dj[bk]; }
    }
  }
  __syncthreads();
  for (int c = tid; c < 400; c += 256)
    out[(long)b * 400 + c] = pathm[c] ? 1.0f : 0.0f;
}

// ---------------------------------------------------------------------------
extern "C" void kernel_launch(void* const* d_in, const int* in_sizes, int n_in,
                              void* d_out, int out_size, void* d_ws, size_t ws_size,
                              hipStream_t stream) {
  const float* d  = (const float*)d_in[0];  // (B, CTX)
  const float* W1 = (const float*)d_in[1];  // (HID, CTX)
  const float* b1 = (const float*)d_in[2];  // (HID)
  const float* W2 = (const float*)d_in[3];  // (M2, HID)
  const float* b2 = (const float*)d_in[4];  // (M2)

  const int B = 1024, CTX = 4096, HID = 8192, M2 = 400;

  // workspace layout (all offsets 256B-aligned): ~100.5 MB total
  char* ws = (char*)d_ws;
  _Float16* W1h = (_Float16*)ws;  ws += (size_t)HID * CTX * sizeof(_Float16);
  _Float16* W2h = (_Float16*)ws;  ws += (size_t)M2  * HID * sizeof(_Float16);
  _Float16* Dh  = (_Float16*)ws;  ws += (size_t)B   * CTX * sizeof(_Float16);
  _Float16* Hh  = (_Float16*)ws;  ws += (size_t)B   * HID * sizeof(_Float16);
  float*    Wg  = (float*)ws;     // B x 400 grid weights

  f32_to_f16_kernel<<<2048, 256, 0, stream>>>(W1, W1h, (long)HID * CTX);
  f32_to_f16_kernel<<<2048, 256, 0, stream>>>(W2, W2h, (long)M2 * HID);
  f32_to_f16_kernel<<<2048, 256, 0, stream>>>(d,  Dh,  (long)B * CTX);

  // H = leaky_relu(D @ W1^T + b1), stored f16
  gemm_f16_wmma<true, true>
      <<<dim3(HID / 128, B / 64), 256, 0, stream>>>(Dh, W1h, b1, (void*)Hh,
                                                    B, HID, CTX);
  // w = H @ W2^T + b2, stored f32
  gemm_f16_wmma<false, false>
      <<<dim3((M2 + 127) / 128, B / 64), 256, 0, stream>>>(Hh, W2h, b2, (void*)Wg,
                                                           B, M2, HID);

  bf_solver<<<B, 256, 0, stream>>>(Wg, (float*)d_out);
}
